// FlaxFullyConnectedTensorProduct_35244501631525
// MI455X (gfx1250) — matrix operand order repaired and checked by reading, not compiled, not run on hardware
//
#include <hip/hip_runtime.h>
#include <hip/hip_fp16.h>

// MI455X / gfx1250, wave32, WMMA f32_16x16x32_f16.
//
// Fused tensor-product GEMM: out[n,w,ch] = sum_{u,v} Aplane_ch[n,(u,v)] * W_ch[(u,v),w]
// M = 4096 (batch), Nw = 128, K = 16384 (u,v).
// - Weights pre-converted to f16 [path][u][w][v] with CG norms folded in.
// - x1/x2 pre-packed as f16 [row][u][{s,x,y,z}] (streamed from L2 by the builder).
// - Workgroup: 128 threads (4 waves), 32 batch rows, one 64-wide w-half.
//   Wave = one 16-w tile x two 16-row subtiles x 4 channels -> 8 accumulators,
//   22 WMMA per K-chunk per wave; B fragments reused across the two row groups.
// - A-plane builder runs in packed f16 (v_perm_b32 + v_pk_mul/fma_f16).

#define INV_SQRT3 0.57735026918962576f
#define INV_SQRT6 0.40824829046386302f

typedef __attribute__((ext_vector_type(16))) _Float16 v16h;
typedef __attribute__((ext_vector_type(8)))  float    v8f;

union V16H { v16h v; float4 f[2]; };
union V8F  { v8f  v; float  f[8]; };
union F4U4 { float4 f; unsigned u[4]; };
union F2U2 { float2 f; unsigned u[2]; };

static __device__ inline __half2 h2perm(unsigned hi, unsigned lo, unsigned sel) {
    unsigned r = __builtin_amdgcn_perm(hi, lo, sel);
    return *(__half2*)&r;
}

// ---------------------------------------------------------------------------
// Pre-pass 1: f32 [u,v,w] weights -> f16 [path][u][w][v], CG norm folded in.
// ---------------------------------------------------------------------------
__global__ __launch_bounds__(256) void prep_weights(
    const float* __restrict__ w0, const float* __restrict__ w1,
    const float* __restrict__ w2, const float* __restrict__ w3,
    const float* __restrict__ w4, __half* __restrict__ dst)
{
    const int p = blockIdx.y;
    const float* src = (p == 0) ? w0 : (p == 1) ? w1 : (p == 2) ? w2 : (p == 3) ? w3 : w4;
    const float scale = (p == 0) ? 1.0f : (p == 4) ? INV_SQRT6 : INV_SQRT3;
    const unsigned i = blockIdx.x * 256u + threadIdx.x;   // < 2^21 per path
    const unsigned v = i & 127u;
    const unsigned w = (i >> 7) & 127u;
    const unsigned u = i >> 14;
    dst[((size_t)p << 21) + i] = __float2half(src[(((u << 7) + v) << 7) + w] * scale);
}

// ---------------------------------------------------------------------------
// Pre-pass 2: pack x1/x2 as f16 [row][u][{s,x,y,z}].
// ---------------------------------------------------------------------------
__global__ __launch_bounds__(256) void prep_x(
    const float* __restrict__ x1_s, const float* __restrict__ x1_v,
    const float* __restrict__ x2_s, const float* __restrict__ x2_v,
    __half* __restrict__ X1p, __half* __restrict__ X2p)
{
    const size_t i = blockIdx.x * 256u + threadIdx.x;     // < 4096*128
    {
        const float  s  = x1_s[i];
        const float* vp = x1_v + i * 3;
        __half2* o = (__half2*)(X1p + i * 4);
        o[0] = __floats2half2_rn(s, vp[0]);
        o[1] = __floats2half2_rn(vp[1], vp[2]);
    }
    {
        const float  s  = x2_s[i];
        const float* vp = x2_v + i * 3;
        __half2* o = (__half2*)(X2p + i * 4);
        o[0] = __floats2half2_rn(s, vp[0]);
        o[1] = __floats2half2_rn(vp[1], vp[2]);
    }
}

// ---------------------------------------------------------------------------
// Main fused kernel.
// ---------------------------------------------------------------------------
__global__ __launch_bounds__(128) void tp_kernel(
    const __half* __restrict__ wt, const __half* __restrict__ X1p,
    const __half* __restrict__ X2p, float* __restrict__ out)
{
    // double-buffered A planes: [buf][plane][row 0..31][32 halves + pad to 40]
    __shared__ __half At[2][11][32][40];   // 56,320 B

    const int tid  = threadIdx.x;          // 0..127
    const int lane = tid & 31;
    const int wv   = tid >> 5;             // 0..3
    const int nb   = blockIdx.x << 5;      // 32-row base
    const int whalf= blockIdx.y;           // 0..1 -> w columns [whalf*64, +64)

    // ---- builder mapping: 512 slots (32 n x 16 v-pairs), 4 slots per thread
    const int bn0 = tid >> 4;              // 0..7 (+8k per slot)
    const int bj  = (tid & 15) << 1;       // 0,2,..,30

    auto buildA = [&](int buf, int uu, int vv0) {
#pragma unroll
        for (int k = 0; k < 4; ++k) {
            const int n   = bn0 + (k << 3);       // 0..31
            const int row = nb + n;
            F2U2 d; d.f = *(const float2*)(X1p + ((size_t)row * 128 + uu) * 4);
            F4U4 q; q.f = *(const float4*)(X2p + ((size_t)row * 128 + vv0 + bj) * 4);
            // x1-side splats: (s1|a0) in d.u[0], (a1|a2) in d.u[1]
            const __half2 s1p = h2perm(d.u[0], d.u[0], 0x01000100u);
            const __half2 a0p = h2perm(d.u[0], d.u[0], 0x03020302u);
            const __half2 a1p = h2perm(d.u[1], d.u[1], 0x01000100u);
            const __half2 a2p = h2perm(d.u[1], d.u[1], 0x03020302u);
            // x2-side (e0,e1) pairs: e0 = (s2|b0),(b1|b2) in q.u[0..1]; e1 in q.u[2..3]
            const __half2 s2p = h2perm(q.u[2], q.u[0], 0x05040100u);
            const __half2 b0p = h2perm(q.u[2], q.u[0], 0x07060302u);
            const __half2 b1p = h2perm(q.u[3], q.u[1], 0x05040100u);
            const __half2 b2p = h2perm(q.u[3], q.u[1], 0x07060302u);

            __half2 P[11];
            P[0]  = __hmul2(s1p, s2p);                                    // 0e x 0e -> 0e
            P[1]  = __hfma2(a0p, b0p, __hfma2(a1p, b1p, __hmul2(a2p, b2p)));  // dot
            P[2]  = __hmul2(s1p, b0p);                                    // 0e x 1o -> 1o
            P[3]  = __hmul2(s1p, b1p);
            P[4]  = __hmul2(s1p, b2p);
            P[5]  = __hmul2(a0p, s2p);                                    // 1o x 0e -> 1o
            P[6]  = __hmul2(a1p, s2p);
            P[7]  = __hmul2(a2p, s2p);
            P[8]  = __hsub2(__hmul2(a1p, b2p), __hmul2(a2p, b1p));        // cross
            P[9]  = __hsub2(__hmul2(a2p, b0p), __hmul2(a0p, b2p));
            P[10] = __hsub2(__hmul2(a0p, b1p), __hmul2(a1p, b0p));
#pragma unroll
            for (int pl = 0; pl < 11; ++pl)
                *(__half2*)&At[buf][pl][n][bj] = P[pl];
        }
    };

    // ---- consumer addressing (ISA 16-bit A 16x32 / B 32x16 layouts)
    const int ar   = lane & 15;               // A row within 16-row group
    const int ac   = (lane < 16) ? 0 : 8;     // A half-offset within row
    const int wl   = lane & 15;               // B column within tile
    const int vsel = (lane < 16) ? 0 : 8;     // B K offset within 16-K block
    const int wcol = (whalf << 6) + (wv << 4) + wl;  // global w column 0..127

    V8F accS[2] = {{}, {}}, accV0[2] = {{}, {}}, accV1[2] = {{}, {}}, accV2[2] = {{}, {}};

    buildA(0, 0, 0);
    __syncthreads();

#define AFRAG(pl, rg, dst)                                                     \
    do {                                                                       \
        (dst).f[0] = *(const float4*)&At[buf][pl][((rg) << 4) + ar][ac];       \
        (dst).f[1] = *(const float4*)&At[buf][pl][((rg) << 4) + ar][ac + 16];  \
    } while (0)
#define BFRAG(p, dst)                                                          \
    do {                                                                       \
        const __half* wp = wt + (((((size_t)(p) << 7) + u) << 7) + wcol) * 128 \
                              + (size_t)(v0 + vsel);                           \
        (dst).f[0] = *(const float4*)(wp);                                     \
        (dst).f[1] = *(const float4*)(wp + 16);                                \
    } while (0)
#define WMMA(a, b, c) \
    (c).v = __builtin_amdgcn_wmma_f32_16x16x32_f16(false, (a).v, false, (b).v, (short)0, (c).v, false, false)

    for (int c = 0; c < 512; ++c) {
        const int buf = c & 1;
        const int u   = c >> 2;
        const int v0  = (c & 3) << 5;

        V16H A0, A1, A2, A3, A4, A5, B0, B1;
        // scalar output: planes 0 (w_sss) and 1 (w_vvs)
        BFRAG(0, B0); BFRAG(1, B1);
        AFRAG(0, 0, A0); AFRAG(0, 1, A1); AFRAG(1, 0, A2); AFRAG(1, 1, A3);
        WMMA(A0, B0, accS[0]); WMMA(A1, B0, accS[1]);
        WMMA(A2, B1, accS[0]); WMMA(A3, B1, accS[1]);
        // vector output, path w_svv (planes 2..4)
        BFRAG(2, B0);
        AFRAG(2, 0, A0); AFRAG(2, 1, A1);
        AFRAG(3, 0, A2); AFRAG(3, 1, A3);
        AFRAG(4, 0, A4); AFRAG(4, 1, A5);
        WMMA(A0, B0, accV0[0]); WMMA(A1, B0, accV0[1]);
        WMMA(A2, B0, accV1[0]); WMMA(A3, B0, accV1[1]);
        WMMA(A4, B0, accV2[0]); WMMA(A5, B0, accV2[1]);
        // vector output, path w_vsv (planes 5..7)
        BFRAG(3, B0);
        AFRAG(5, 0, A0); AFRAG(5, 1, A1);
        AFRAG(6, 0, A2); AFRAG(6, 1, A3);
        AFRAG(7, 0, A4); AFRAG(7, 1, A5);
        WMMA(A0, B0, accV0[0]); WMMA(A1, B0, accV0[1]);
        WMMA(A2, B0, accV1[0]); WMMA(A3, B0, accV1[1]);
        WMMA(A4, B0, accV2[0]); WMMA(A5, B0, accV2[1]);
        // vector output, path w_vvv / cross (planes 8..10)
        BFRAG(4, B0);
        AFRAG(8, 0, A0); AFRAG(8, 1, A1);
        AFRAG(9, 0, A2); AFRAG(9, 1, A3);
        AFRAG(10, 0, A4); AFRAG(10, 1, A5);
        WMMA(A0, B0, accV0[0]); WMMA(A1, B0, accV0[1]);
        WMMA(A2, B0, accV1[0]); WMMA(A3, B0, accV1[1]);
        WMMA(A4, B0, accV2[0]); WMMA(A5, B0, accV2[1]);

        if (c + 1 < 512)
            buildA((c + 1) & 1, (c + 1) >> 2, ((c + 1) & 3) << 5);
        __syncthreads();
    }

    // ---- writeback: C/D layout = VGPR r -> M = r + (lane<16 ? 0 : 8), N = lane%16
    const int mh = (lane < 16) ? 0 : 8;
#pragma unroll
    for (int rg = 0; rg < 2; ++rg) {
#pragma unroll
        for (int r = 0; r < 8; ++r) {
            const int row = nb + (rg << 4) + r + mh;
            float* orow = out + (size_t)row * 512;
            orow[wcol]               = accS[rg].f[r];
            orow[128 + wcol * 3 + 0] = accV0[rg].f[r];
            orow[128 + wcol * 3 + 1] = accV1[rg].f[r];
            orow[128 + wcol * 3 + 2] = accV2[rg].f[r];
        }
    }
}

// ---------------------------------------------------------------------------
extern "C" void kernel_launch(void* const* d_in, const int* in_sizes, int n_in,
                              void* d_out, int out_size, void* d_ws, size_t ws_size,
                              hipStream_t stream)
{
    (void)in_sizes; (void)n_in; (void)out_size; (void)ws_size;

    __half* wt  = (__half*)d_ws;                 // 5 * 2^21 halves = 20 MiB
    __half* X1p = wt + (5u << 21);               // 4096*128*4 halves = 4 MiB
    __half* X2p = X1p + (4096u * 128u * 4u);     // 4 MiB   (total 28 MiB scratch)

    dim3 pgrid(8192, 5);
    prep_weights<<<pgrid, 256, 0, stream>>>(
        (const float*)d_in[4], (const float*)d_in[5], (const float*)d_in[6],
        (const float*)d_in[7], (const float*)d_in[8], wt);

    prep_x<<<(4096 * 128) / 256, 256, 0, stream>>>(
        (const float*)d_in[0], (const float*)d_in[1],
        (const float*)d_in[2], (const float*)d_in[3], X1p, X2p);

    dim3 mgrid(4096 / 32, 2);                    // 128 row-blocks x 2 w-halves
    tp_kernel<<<mgrid, 128, 0, stream>>>(wt, X1p, X2p, (float*)d_out);
}
